// MultiHeadSelfAttention_63359357551195
// MI455X (gfx1250) — matrix operand order repaired and checked by reading, not compiled, not run on hardware
//
#include <hip/hip_runtime.h>

// ---------------- types ----------------
typedef unsigned short u16;
typedef unsigned int   u32;
typedef unsigned long long u64;
typedef __attribute__((ext_vector_type(16))) __bf16 v16bf;
typedef __attribute__((ext_vector_type(8)))  float  v8f;

union FragAB { v16bf v; uint4 q[2]; };   // 32-byte WMMA A/B operand

__device__ __forceinline__ u16 f32_bf16(float f) {
  u32 u = __builtin_bit_cast(u32, f);
  u32 r = u + 0x7FFFu + ((u >> 16) & 1u);   // round-to-nearest-even
  return (u16)(r >> 16);
}

__device__ __forceinline__ v8f vzero8() {
  v8f z = {0.f,0.f,0.f,0.f,0.f,0.f,0.f,0.f};
  return z;
}

// CDNA5 async global->LDS copy (ASYNCcnt): VGLOBAL "GV" form.
// lds  = byte offset within the workgroup LDS allocation (addr[31:0] of generic ptr)
// gptr = 64-bit global address in a VGPR pair
__device__ __forceinline__ void async_ld_b128(u32 lds, const u16* gptr) {
  asm volatile("global_load_async_to_lds_b128 %0, %1, off"
               :: "v"(lds), "v"((u64)(size_t)gptr) : "memory");
}
__device__ __forceinline__ void wait_async0() {
  asm volatile("s_wait_asynccnt 0x0" ::: "memory");
}
__device__ __forceinline__ u32 lds_off(const void* p) {
  return (u32)(size_t)p;   // flat->LDS mapping uses addr[31:0]
}

// problem constants
#define BATCH 4
#define SEQ   2048
#define CDIM  1024
#define NH    16
#define DH    64
#define BT    (BATCH*SEQ)     // 8192

// ---------------- conversion kernels ----------------
__global__ __launch_bounds__(256) void cvt_x_kernel(const float* __restrict__ x,
                                                    u16* __restrict__ xb) {
  int i = (blockIdx.x * 256 + threadIdx.x) * 4;
  float4 f = *(const float4*)(x + i);
  u32 lo = (u32)f32_bf16(f.x) | ((u32)f32_bf16(f.y) << 16);
  u32 hi = (u32)f32_bf16(f.z) | ((u32)f32_bf16(f.w) << 16);
  uint2 p; p.x = lo; p.y = hi;
  *(uint2*)(xb + i) = p;
}

// w: (CDIM, N) fp32 row-major  ->  wT: (N, CDIM) bf16 row-major
__global__ __launch_bounds__(256) void cvt_wT_kernel(const float* __restrict__ w,
                                                     u16* __restrict__ wT, int N) {
  int i = blockIdx.x * 256 + threadIdx.x;   // i = n*CDIM + c
  int n = i >> 10;
  int c = i & 1023;
  wT[i] = f32_bf16(w[(size_t)c * N + n]);
}

// ---------------- shared GEMM core ----------------
// Block tile 128(M) x 128(N), K-step 64, double-buffered async LDS staging.
// 8 waves arranged 4(M) x 2(N); wave tile 32x64 = 2x4 WMMA accumulators.
// 16 WMMA per wave per K-step vs 24 ds_load_b128 fragment loads.
#define LDT 72   // padded bf16 row stride (144B: 16B aligned, conflict-free cycle)

__device__ __forceinline__ void stage_tile32(u16* ldsDst, const u16* gSrc) {
  // one thread stages 32 bf16 (64B) = 4 async b128 transfers
#pragma unroll
  for (int j = 0; j < 4; ++j)
    async_ld_b128(lds_off(ldsDst + j * 8), gSrc + j * 8);
}

__device__ __forceinline__ void gemm_compute_step(const u16* bufA, const u16* bufB,
                                                  v8f (&acc)[2][4],
                                                  int wm, int wn, int l15, bool hiL) {
#pragma unroll
  for (int kc = 0; kc < 2; ++kc) {          // two 32-wide K chunks of the 64 slab
    FragAB aF[2], bF[4];
#pragma unroll
    for (int mt = 0; mt < 2; ++mt) {
      const u16* p = bufA + (wm * 32 + mt * 16 + l15) * LDT + kc * 32 + (hiL ? 8 : 0);
      aF[mt].q[0] = *(const uint4*)p;
      aF[mt].q[1] = *(const uint4*)(p + 16);
    }
#pragma unroll
    for (int nt = 0; nt < 4; ++nt) {
      const u16* p = bufB + (wn * 64 + nt * 16 + l15) * LDT + kc * 32 + (hiL ? 8 : 0);
      bF[nt].q[0] = *(const uint4*)p;
      bF[nt].q[1] = *(const uint4*)(p + 16);
    }
#pragma unroll
    for (int mt = 0; mt < 2; ++mt)
#pragma unroll
      for (int nt = 0; nt < 4; ++nt)
        acc[mt][nt] = __builtin_amdgcn_wmma_f32_16x16x32_bf16(
            false, aF[mt].v, false, bF[nt].v, (short)0, acc[mt][nt], false, false);
  }
}

__device__ __forceinline__ void gemm_core(const u16* __restrict__ A,
                                          const u16* __restrict__ Bt,
                                          int K, int mBase, int nBase,
                                          u16* sA0, u16* sA1, u16* sB0, u16* sB1,
                                          v8f (&acc)[2][4], int tid) {
  const int lane = tid & 31;
  const int wid  = tid >> 5;
  const int wm   = wid & 3;          // 32-row slab
  const int wn   = wid >> 2;         // 64-col slab
  const int l15  = lane & 15;
  const bool hiL = (lane & 16) != 0;

  // staging split: thread -> (row 0..127, col half 0/32)
  const int rowS = tid >> 1;
  const int colS = (tid & 1) * 32;
  const u16* gA = A  + (size_t)(mBase + rowS) * K + colS;
  const u16* gB = Bt + (size_t)(nBase + rowS) * K + colS;
  const int ldsRow = rowS * LDT + colS;

  const int nsteps = K / 64;
  // prologue: stage slab 0 into buffer 0
  stage_tile32(sA0 + ldsRow, gA);
  stage_tile32(sB0 + ldsRow, gB);

  for (int i = 0; i < nsteps; ++i) {
    u16* curA = (i & 1) ? sA1 : sA0;
    u16* curB = (i & 1) ? sB1 : sB0;
    u16* nxtA = (i & 1) ? sA0 : sA1;
    u16* nxtB = (i & 1) ? sB0 : sB1;

    wait_async0();          // own async stores to LDS done
    __syncthreads();        // whole block's staging visible

    if (i + 1 < nsteps) {   // async prefetch of next slab; overlaps with WMMA below
      stage_tile32(nxtA + ldsRow, gA + (i + 1) * 64);
      stage_tile32(nxtB + ldsRow, gB + (i + 1) * 64);
    }

    gemm_compute_step(curA, curB, acc, wm, wn, l15, hiL);
    __syncthreads();        // everyone done reading cur before it becomes a prefetch target
  }
}

// ---------------- qkv GEMM: xb(8192,1024) @ Wqkv -> Q/K/Vt head layouts ----------------
__global__ __launch_bounds__(256) void gemm_qkv_kernel(const u16* __restrict__ xb,
                                                       const u16* __restrict__ wT,
                                                       u16* __restrict__ qb,
                                                       u16* __restrict__ kb,
                                                       u16* __restrict__ vtb) {
  __shared__ u16 sA0[128 * LDT];
  __shared__ u16 sA1[128 * LDT];
  __shared__ u16 sB0[128 * LDT];
  __shared__ u16 sB1[128 * LDT];
  const int tid = threadIdx.x;
  const int mBase = blockIdx.y * 128;
  const int nBase = blockIdx.x * 128;

  v8f acc[2][4];
#pragma unroll
  for (int i = 0; i < 2; ++i)
#pragma unroll
    for (int j = 0; j < 4; ++j) acc[i][j] = vzero8();

  gemm_core(xb, wT, CDIM, mBase, nBase, sA0, sA1, sB0, sB1, acc, tid);

  const int lane = tid & 31, wid = tid >> 5;
  const int wm = wid & 3, wn = wid >> 2;
  const int l15 = lane & 15;
  const bool hiL = (lane & 16) != 0;
#pragma unroll
  for (int mt = 0; mt < 2; ++mt)
#pragma unroll
    for (int nt = 0; nt < 4; ++nt)
#pragma unroll
      for (int r = 0; r < 8; ++r) {
        int row = mBase + wm * 32 + mt * 16 + r + (hiL ? 8 : 0);   // token index in BT
        int col = nBase + wn * 64 + nt * 16 + l15;                 // 0..3071
        float v = acc[mt][nt][r];
        int b = row >> 11, t = row & 2047;
        int which = col >> 10, f = col & 1023;
        int h = f >> 6, d = f & 63;
        int bh = b * NH + h;
        if (which == 0)
          qb[((size_t)bh * SEQ + t) * DH + d] = f32_bf16(v * 0.125f);  // pre-scale 1/sqrt(64)
        else if (which == 1)
          kb[((size_t)bh * SEQ + t) * DH + d] = f32_bf16(v);
        else
          vtb[((size_t)bh * DH + d) * SEQ + t] = f32_bf16(v);          // V transposed (Dh,T)
      }
}

// ---------------- flash attention: one block per (head, 128 queries) ----------------
__global__ __launch_bounds__(256) void attn_kernel(const u16* __restrict__ qb,
                                                   const u16* __restrict__ kb,
                                                   const u16* __restrict__ vtb,
                                                   u16* __restrict__ ob) {
  const int tid  = threadIdx.x;
  const int lane = tid & 31;
  const int wid  = tid >> 5;
  const int l15  = lane & 15;
  const bool hiL = (lane & 16) != 0;
  const int bh = blockIdx.y;
  const int t0 = blockIdx.x * 128;
  const int rowBase = t0 + wid * 16;        // this wave's 16 query rows

  const u16* qB = qb  + (size_t)bh * SEQ * DH;
  const u16* kB = kb  + (size_t)bh * SEQ * DH;
  const u16* vB = vtb + (size_t)bh * DH * SEQ;

  __shared__ u16 ldsP[8][16][72];           // per-wave P tile (16x64 bf16, padded)

  // Q A-fragments (two 32-wide K chunks covering Dh=64), loaded once
  FragAB qF[2];
  {
    const u16* base = qB + (size_t)(rowBase + l15) * DH + (hiL ? 8 : 0);
#pragma unroll
    for (int kk = 0; kk < 2; ++kk) {
      qF[kk].q[0] = *(const uint4*)(base + kk * 32);
      qF[kk].q[1] = *(const uint4*)(base + kk * 32 + 16);
    }
  }

  v8f accO[4];
#pragma unroll
  for (int dn = 0; dn < 4; ++dn) accO[dn] = vzero8();
  float mrow[8], lrow[8];
#pragma unroll
  for (int r = 0; r < 8; ++r) { mrow[r] = -3.0e38f; lrow[r] = 0.f; }

  for (int tk0 = 0; tk0 < t0 + 128; tk0 += 64) {       // uniform trip count per block
    // ---- S = Q @ K^T (16 x 64 per wave) ----
    v8f sAcc[4];
#pragma unroll
    for (int nt = 0; nt < 4; ++nt) sAcc[nt] = vzero8();
#pragma unroll
    for (int nt = 0; nt < 4; ++nt)
#pragma unroll
      for (int kk = 0; kk < 2; ++kk) {
        FragAB kF;
        const u16* kp = kB + (size_t)(tk0 + nt * 16 + l15) * DH + kk * 32 + (hiL ? 8 : 0);
        kF.q[0] = *(const uint4*)kp;
        kF.q[1] = *(const uint4*)(kp + 16);
        sAcc[nt] = __builtin_amdgcn_wmma_f32_16x16x32_bf16(
            false, qF[kk].v, false, kF.v, (short)0, sAcc[nt], false, false);
      }

    // ---- causal mask + row max ----
    float mx[8];
#pragma unroll
    for (int r = 0; r < 8; ++r) mx[r] = -3.0e38f;
#pragma unroll
    for (int nt = 0; nt < 4; ++nt)
#pragma unroll
      for (int r = 0; r < 8; ++r) {
        int col = tk0 + nt * 16 + l15;
        int row = rowBase + r + (hiL ? 8 : 0);
        float s = sAcc[nt][r];
        if (col > row) s = -3.0e38f;
        sAcc[nt][r] = s;
        mx[r] = fmaxf(mx[r], s);
      }
#pragma unroll
    for (int r = 0; r < 8; ++r) {           // reduce across the 16 lanes of each half
      float v = mx[r];
      v = fmaxf(v, __shfl_xor(v, 1, 32));
      v = fmaxf(v, __shfl_xor(v, 2, 32));
      v = fmaxf(v, __shfl_xor(v, 4, 32));
      v = fmaxf(v, __shfl_xor(v, 8, 32));
      mx[r] = v;
    }

    // ---- online softmax update ----
    float scale[8];
#pragma unroll
    for (int r = 0; r < 8; ++r) {
      float nm = fmaxf(mrow[r], mx[r]);
      scale[r] = __expf(mrow[r] - nm);
      mrow[r] = nm;
    }
    float rs[8];
#pragma unroll
    for (int r = 0; r < 8; ++r) rs[r] = 0.f;
#pragma unroll
    for (int nt = 0; nt < 4; ++nt)
#pragma unroll
      for (int r = 0; r < 8; ++r) {
        float p = __expf(sAcc[nt][r] - mrow[r]);
        sAcc[nt][r] = p;
        rs[r] += p;
      }
#pragma unroll
    for (int r = 0; r < 8; ++r) {
      float v = rs[r];
      v += __shfl_xor(v, 1, 32);
      v += __shfl_xor(v, 2, 32);
      v += __shfl_xor(v, 4, 32);
      v += __shfl_xor(v, 8, 32);
      lrow[r] = lrow[r] * scale[r] + v;
    }
#pragma unroll
    for (int dn = 0; dn < 4; ++dn)
#pragma unroll
      for (int r = 0; r < 8; ++r) accO[dn][r] *= scale[r];

    // ---- P (C/D layout) -> LDS -> A-fragment layout ----
#pragma unroll
    for (int nt = 0; nt < 4; ++nt)
#pragma unroll
      for (int r = 0; r < 8; ++r)
        ldsP[wid][r + (hiL ? 8 : 0)][nt * 16 + l15] = f32_bf16(sAcc[nt][r]);
    __syncthreads();

    FragAB pF[2];
    {
      const u16* pbase = &ldsP[wid][l15][hiL ? 8 : 0];
#pragma unroll
      for (int kk = 0; kk < 2; ++kk) {
        pF[kk].q[0] = *(const uint4*)(pbase + kk * 32);
        pF[kk].q[1] = *(const uint4*)(pbase + kk * 32 + 16);
      }
    }

    // ---- O += P @ V  (V stored transposed (Dh,T) so B-frags are contiguous) ----
#pragma unroll
    for (int dn = 0; dn < 4; ++dn)
#pragma unroll
      for (int kk = 0; kk < 2; ++kk) {
        FragAB vF;
        const u16* vp = vB + (size_t)(dn * 16 + l15) * SEQ + tk0 + kk * 32 + (hiL ? 8 : 0);
        vF.q[0] = *(const uint4*)vp;
        vF.q[1] = *(const uint4*)(vp + 16);
        accO[dn] = __builtin_amdgcn_wmma_f32_16x16x32_bf16(
            false, pF[kk].v, false, vF.v, (short)0, accO[dn], false, false);
      }
    __syncthreads();
  }

  // ---- normalize + write O in (B,T,C) bf16 ----
  const int b = bh >> 4, h = bh & 15;
  u16* oB = ob + ((size_t)b * SEQ) * CDIM + h * DH;
#pragma unroll
  for (int dn = 0; dn < 4; ++dn)
#pragma unroll
    for (int r = 0; r < 8; ++r) {
      int row = rowBase + r + (hiL ? 8 : 0);
      int d = dn * 16 + l15;
      oB[(size_t)row * CDIM + d] = f32_bf16(accO[dn][r] / lrow[r]);
    }
}

// ---------------- projection GEMM: ob(8192,1024) @ Wproj -> fp32 out ----------------
__global__ __launch_bounds__(256) void gemm_proj_kernel(const u16* __restrict__ ob,
                                                        const u16* __restrict__ wT,
                                                        float* __restrict__ out) {
  __shared__ u16 sA0[128 * LDT];
  __shared__ u16 sA1[128 * LDT];
  __shared__ u16 sB0[128 * LDT];
  __shared__ u16 sB1[128 * LDT];
  const int tid = threadIdx.x;
  const int mBase = blockIdx.y * 128;
  const int nBase = blockIdx.x * 128;

  v8f acc[2][4];
#pragma unroll
  for (int i = 0; i < 2; ++i)
#pragma unroll
    for (int j = 0; j < 4; ++j) acc[i][j] = vzero8();

  gemm_core(ob, wT, CDIM, mBase, nBase, sA0, sA1, sB0, sB1, acc, tid);

  const int lane = tid & 31, wid = tid >> 5;
  const int wm = wid & 3, wn = wid >> 2;
  const int l15 = lane & 15;
  const bool hiL = (lane & 16) != 0;
#pragma unroll
  for (int mt = 0; mt < 2; ++mt)
#pragma unroll
    for (int nt = 0; nt < 4; ++nt)
#pragma unroll
      for (int r = 0; r < 8; ++r) {
        int row = mBase + wm * 32 + mt * 16 + r + (hiL ? 8 : 0);
        int col = nBase + wn * 64 + nt * 16 + l15;
        out[(size_t)row * CDIM + col] = acc[mt][nt][r];
      }
}

// ---------------- launch ----------------
extern "C" void kernel_launch(void* const* d_in, const int* in_sizes, int n_in,
                              void* d_out, int out_size, void* d_ws, size_t ws_size,
                              hipStream_t stream) {
  (void)in_sizes; (void)n_in; (void)out_size; (void)ws_size;
  const float* x      = (const float*)d_in[0];   // (4,2048,1024)
  const float* w_qkv  = (const float*)d_in[1];   // (1024,3072)
  const float* w_proj = (const float*)d_in[2];   // (1024,1024)
  float* out = (float*)d_out;                    // (4,2048,1024)

  // workspace carve-up (bf16 element counts)
  u16* xb    = (u16*)d_ws;                 // 8M  : x in bf16
  u16* wqkvT = xb    + (size_t)BT * CDIM;  // 3M  : Wqkv^T (3072,1024)
  u16* wprojT= wqkvT + (size_t)3 * CDIM * CDIM;   // 1M : Wproj^T
  u16* qb    = wprojT+ (size_t)CDIM * CDIM;       // 8M : (B,H,T,Dh), pre-scaled
  u16* kb    = qb    + (size_t)BT * CDIM;         // 8M : (B,H,T,Dh)
  u16* vtb   = kb    + (size_t)BT * CDIM;         // 8M : (B,H,Dh,T)
  u16* ob    = vtb   + (size_t)BT * CDIM;         // 8M : attn out (B,T,C)

  cvt_x_kernel<<<(BT * CDIM) / (256 * 4), 256, 0, stream>>>(x, xb);
  cvt_wT_kernel<<<(3 * CDIM * CDIM) / 256, 256, 0, stream>>>(w_qkv, wqkvT, 3 * CDIM);
  cvt_wT_kernel<<<(CDIM * CDIM) / 256, 256, 0, stream>>>(w_proj, wprojT, CDIM);

  gemm_qkv_kernel<<<dim3((3 * CDIM) / 128, BT / 128), 256, 0, stream>>>(xb, wqkvT, qb, kb, vtb);
  attn_kernel<<<dim3(SEQ / 128, BATCH * NH), 256, 0, stream>>>(qb, kb, vtb, ob);
  gemm_proj_kernel<<<dim3(CDIM / 128, BT / 128), 256, 0, stream>>>(ob, wprojT, out);
}
